// MogLSTM_59339268162241
// MI455X (gfx1250) — compile-verified, matrix operand order checked by reading, not run on hardware
//
#include <hip/hip_runtime.h>

// ---------------------------------------------------------------------------
// Mogrifier-LSTM (2 layers, SEQ=512, B=64, I=H=512) for MI455X / gfx1250.
// Batch-parallel persistent kernel: 4 workgroups (one per 16-row batch slice)
// x 1024 threads (32 waves). All matmuls are M=16 WMMA tiles:
//   mogrify:  (16x512)*(512x512)   -> 32 N-tiles, 1 per wave, 2 acc chains
//   gates:    (16x512)*(512x2048)  -> 128 N-tiles, 4 per wave, 4 acc chains
// K-loop: 16 chunks of 32 -> v_wmma_f32_16x16x32_bf16, f32 accumulation.
// Weights pre-converted to bf16 and pre-swizzled into per-lane WMMA B-fragment
// order by a prep kernel; streamed from L2 (13 MB << 192 MB L2).
// input_seq pre-converted to bf16 (ws permitting); x_{t+1} register-prefetched.
// Activations in LDS (bf16); cell state c in VGPRs; biases hoisted to VGPRs.
// Elementwise gates use single-instruction v_rcp_f32/v_exp_f32(/v_tanh_f32).
// ---------------------------------------------------------------------------

typedef __attribute__((ext_vector_type(16))) __bf16 v16bf;
typedef __attribute__((ext_vector_type(8)))  __bf16 v8bf;
typedef __attribute__((ext_vector_type(8)))  float  v8f;

constexpr int SEQ   = 512;
constexpr int BATCH = 64;
constexpr int HD    = 512;   // I == H == 512
constexpr int NSTEP = 5;
constexpr int KCN   = 16;    // K chunks of 32 (512/32)
constexpr int SP    = 544;   // LDS row stride in bf16 elems (padded, 16B aligned)
constexpr int TILE_ELEMS = KCN * 512;  // bf16 elems per packed 16-wide N-tile

// ---- scalar helpers -------------------------------------------------------

__device__ __forceinline__ float bf2f(__bf16 b) {
  unsigned short s = __builtin_bit_cast(unsigned short, b);
  unsigned u = ((unsigned)s) << 16;
  return __builtin_bit_cast(float, u);
}

__device__ __forceinline__ __bf16 f2bf(float f) {
  unsigned u = __builtin_bit_cast(unsigned, f);
  u += 0x7FFFu + ((u >> 16) & 1u);           // round-to-nearest-even
  unsigned short s = (unsigned short)(u >> 16);
  return __builtin_bit_cast(__bf16, s);
}

// Fast transcendentals: single-instruction TRANS ops (co-execute with WMMA).
__device__ __forceinline__ float fast_rcp(float x) {
#if __has_builtin(__builtin_amdgcn_rcpf)
  return __builtin_amdgcn_rcpf(x);
#else
  return 1.0f / x;
#endif
}

__device__ __forceinline__ float fast_exp2(float x) {
#if __has_builtin(__builtin_amdgcn_exp2f)
  return __builtin_amdgcn_exp2f(x);
#else
  return __expf(x * 0.6931471805599453f);
#endif
}

__device__ __forceinline__ float sigm(float z) {
  return fast_rcp(1.0f + fast_exp2(-1.4426950408889634f * z));
}

__device__ __forceinline__ float fast_tanh(float z) {
#if __has_builtin(__builtin_amdgcn_tanhf)
  return __builtin_amdgcn_tanhf(z);   // v_tanh_f32 (CDNA5 TRANS op)
#else
  float e = fast_exp2(2.8853900817779268f * z);   // exp(2z)
  return 1.0f - 2.0f * fast_rcp(e + 1.0f);
#endif
}

// ---- weight pre-pack: f32 row-major (N x K) -> bf16 WMMA-B fragment order --
// Fragment (nt, kc) = 32 lanes x 16 bf16 (32B/lane). B[k][n] = W[n][k].
//   lanes 0-15 : N = lane,    elems 0-7 -> K {0..7},  elems 8-15 -> K {16..23}
//   lanes16-31 : N = lane-16, elems 0-7 -> K {8..15}, elems 8-15 -> K {24..31}

__global__ void pack_w_kernel(const float* __restrict__ src,
                              __bf16* __restrict__ dst,
                              int nrows, int ncols) {
  int idx = blockIdx.x * blockDim.x + threadIdx.x;
  int total = nrows * ncols;
  if (idx >= total) return;
  int frag = idx >> 9;            // 512 elems per fragment
  int le   = idx & 511;
  int lane = le >> 4;
  int e    = le & 15;
  int kcc  = ncols >> 5;          // K chunks per row (== 16 here)
  int nt   = frag / kcc;
  int kc   = frag - nt * kcc;
  int n    = (nt << 4) + (lane & 15);
  int kk   = (lane < 16) ? ((e < 8) ? e : e + 8)
                         : ((e < 8) ? e + 8 : e + 16);
  int k    = (kc << 5) + kk;
  dst[idx] = f2bf(src[(size_t)n * ncols + k]);
}

// ---- input pre-convert: f32 -> bf16 (same layout) -------------------------

__global__ void cvt_x_kernel(const float* __restrict__ src,
                             __bf16* __restrict__ dst, int total4) {
  int i = blockIdx.x * blockDim.x + threadIdx.x;
  if (i >= total4) return;
  const float4 f = *(const float4*)(src + (size_t)i * 4);
  __bf16* p = dst + (size_t)i * 4;
  p[0] = f2bf(f.x); p[1] = f2bf(f.y); p[2] = f2bf(f.z); p[3] = f2bf(f.w);
}

// ---- WMMA micro-kernels ---------------------------------------------------
// A fragment (16x32 bf16) from LDS activations (rows = batch, cols = K):
//   lanes 0-15 : row = lane,    K {k0..k0+7} and {k0+16..k0+23}
//   lanes16-31 : row = lane-16, K {k0+8..k0+15} and {k0+24..k0+31}

__device__ __forceinline__ v16bf load_a_frag(const __bf16* __restrict__ aLds,
                                             int lane, int kc) {
  int row = lane & 15;
  int off = row * SP + (kc << 5) + ((lane >> 4) << 3);
  union { v16bf v; v8bf h[2]; } A;
  A.h[0] = *(const v8bf*)(aLds + off);        // ds_load_b128
  A.h[1] = *(const v8bf*)(aLds + off + 16);   // ds_load_b128
  return A.v;
}

// Two independent accumulator chains over even/odd K-chunks: lets the wave
// issue WMMAs back-to-back instead of serializing on one accumulator.
__device__ __forceinline__ v8f mm_tile(const __bf16* __restrict__ aLds,
                                       const __bf16* __restrict__ bTile,
                                       int lane) {
  v8f a0 = {0.f, 0.f, 0.f, 0.f, 0.f, 0.f, 0.f, 0.f};
  v8f a1 = {0.f, 0.f, 0.f, 0.f, 0.f, 0.f, 0.f, 0.f};
#pragma unroll 4
  for (int kc = 0; kc < KCN; kc += 2) {
    v16bf A0 = load_a_frag(aLds, lane, kc);
    v16bf B0 = *(const v16bf*)(bTile + kc * 512 + lane * 16);
    v16bf A1 = load_a_frag(aLds, lane, kc + 1);
    v16bf B1 = *(const v16bf*)(bTile + (kc + 1) * 512 + lane * 16);
    a0 = __builtin_amdgcn_wmma_f32_16x16x32_bf16(
        false, A0, false, B0, (short)0, a0, false, false);
    a1 = __builtin_amdgcn_wmma_f32_16x16x32_bf16(
        false, A1, false, B1, (short)0, a1, false, false);
  }
  return a0 + a1;
}

__device__ __forceinline__ void mm_gates(const __bf16* __restrict__ aLds,
                                         const __bf16* __restrict__ wBase,
                                         int wv, int lane, v8f acc[4]) {
#pragma unroll 2
  for (int kc = 0; kc < KCN; ++kc) {
    v16bf A = load_a_frag(aLds, lane, kc);
#pragma unroll
    for (int g = 0; g < 4; ++g) {
      const __bf16* bt = wBase + (size_t)(g * 32 + wv) * TILE_ELEMS
                       + kc * 512 + lane * 16;
      v16bf B = *(const v16bf*)bt;
      acc[g] = __builtin_amdgcn_wmma_f32_16x16x32_bf16(
          false, A, false, B, (short)0, acc[g], false, false);
    }
  }
}

// ---- one Mogrifier-LSTM cell (16-row batch slice, whole workgroup) --------

__device__ __forceinline__ void mog_cell(
    __bf16* __restrict__ xls,     // layer input, mogrified in place
    __bf16* __restrict__ hls,     // recurrent h, mogrified + overwritten
    v8f& cacc,                    // this wave's 16x16 tile of c (VGPRs)
    const __bf16* __restrict__ mogp,
    const __bf16* __restrict__ wihp, const __bf16* __restrict__ whhp,
    const float (&mb)[NSTEP], const float (&gb)[4],
    int wv, int lane,
    float* __restrict__ outT,     // nullptr, or out + t*B*H (layer 2)
    int m0) {
  const int col   = (wv << 4) + (lane & 15);
  const int rbase = (lane >> 4) << 3;

  // ---- 5 mogrifier phases: even i updates x (reads h), odd updates h ----
#pragma unroll
  for (int i = 0; i < NSTEP; ++i) {
    __bf16* src = (i & 1) ? xls : hls;
    __bf16* dst = (i & 1) ? hls : xls;
    v8f acc = mm_tile(src, mogp + (size_t)(i * 32 + wv) * TILE_ELEMS, lane);
    const float b = mb[i];
#pragma unroll
    for (int r = 0; r < 8; ++r) {
      int row = rbase + r;
      float gate = 2.0f * sigm(acc[r] + b);
      float xo = bf2f(dst[row * SP + col]);
      dst[row * SP + col] = f2bf(gate * xo);
    }
    __syncthreads();
  }

  // ---- LSTM gates: x@Wih.T + h@Whh.T + biases (gate order i,f,g,o) ----
  v8f acc4[4];
#pragma unroll
  for (int g = 0; g < 4; ++g)
    acc4[g] = (v8f){0.f, 0.f, 0.f, 0.f, 0.f, 0.f, 0.f, 0.f};
  mm_gates(xls, wihp, wv, lane, acc4);
  mm_gates(hls, whhp, wv, lane, acc4);

  __syncthreads();  // all waves done reading hls before we overwrite it

#pragma unroll
  for (int r = 0; r < 8; ++r) {
    int row = rbase + r;
    float ig = sigm(acc4[0][r] + gb[0]);
    float fg = sigm(acc4[1][r] + gb[1]);
    float gg = fast_tanh(acc4[2][r] + gb[2]);
    float og = sigm(acc4[3][r] + gb[3]);
    float cv = fg * cacc[r] + ig * gg;
    cacc[r] = cv;
    float hv = og * fast_tanh(cv);
    hls[row * SP + col] = f2bf(hv);
    if (outT) outT[(size_t)(m0 + row) * HD + col] = hv;
  }
  __syncthreads();
}

// ---- persistent main kernel: 1 block per 16-row batch group ---------------
// XBF16: input_seq pre-converted to bf16 in ws (enables uint4 staging +
// register prefetch of x_{t+1} behind the current step's compute).

template <bool XBF16>
__global__ void __launch_bounds__(1024, 1)
moglstm_main(const float* __restrict__ input_seq,
             const __bf16* __restrict__ xseq_bf,
             const float* __restrict__ l1_mogb, const float* __restrict__ l1_bih,
             const float* __restrict__ l1_bhh,
             const float* __restrict__ l2_mogb, const float* __restrict__ l2_bih,
             const float* __restrict__ l2_bhh,
             const __bf16* __restrict__ l1_mogp, const __bf16* __restrict__ l1_wihp,
             const __bf16* __restrict__ l1_whhp,
             const __bf16* __restrict__ l2_mogp, const __bf16* __restrict__ l2_wihp,
             const __bf16* __restrict__ l2_whhp,
             float* __restrict__ out) {
  __shared__ __align__(16) __bf16 xs[16 * SP];
  __shared__ __align__(16) __bf16 h1s[16 * SP];
  __shared__ __align__(16) __bf16 h2s[16 * SP];

  const int tid  = threadIdx.x;
  const int wv   = tid >> 5;
  const int lane = tid & 31;
  const int m0   = blockIdx.x << 4;   // batch-row base for this workgroup
  const int col  = (wv << 4) + (lane & 15);

  // hoist all biases for this lane's columns into registers (whole sequence)
  float mb1[NSTEP], mb2[NSTEP];
#pragma unroll
  for (int i = 0; i < NSTEP; ++i) {
    mb1[i] = l1_mogb[i * HD + col];
    mb2[i] = l2_mogb[i * HD + col];
  }
  float gb1[4], gb2[4];
#pragma unroll
  for (int g = 0; g < 4; ++g) {
    gb1[g] = l1_bih[g * HD + col] + l1_bhh[g * HD + col];
    gb2[g] = l2_bih[g * HD + col] + l2_bhh[g * HD + col];
  }

  // zero-init recurrent state
  for (int i = tid; i < 16 * SP; i += 1024) {
    h1s[i] = f2bf(0.f);
    h2s[i] = f2bf(0.f);
  }
  v8f c1 = {0.f, 0.f, 0.f, 0.f, 0.f, 0.f, 0.f, 0.f};
  v8f c2 = {0.f, 0.f, 0.f, 0.f, 0.f, 0.f, 0.f, 0.f};

  // per-thread staging slice of x_t (8 elems): prefetched into registers
  const int xrow8 = tid >> 6, xc8 = (tid & 63) << 3;   // bf16 path: 1x uint4
  const int xrowA = tid >> 7, xc4a = (tid & 127) << 2; // f32 path: 2x float4
  const int xrowB = (tid + 1024) >> 7, xc4b = ((tid + 1024) & 127) << 2;

  uint4  xrb;
  float4 xrfA, xrfB;
  auto fetch_x = [&](int t) {
    if (XBF16) {
      xrb = *(const uint4*)(xseq_bf + ((size_t)t * BATCH + m0 + xrow8) * HD + xc8);
    } else {
      xrfA = *(const float4*)(input_seq + ((size_t)t * BATCH + m0 + xrowA) * HD + xc4a);
      xrfB = *(const float4*)(input_seq + ((size_t)t * BATCH + m0 + xrowB) * HD + xc4b);
    }
  };
  auto store_x = [&]() {
    if (XBF16) {
      *(uint4*)(xs + xrow8 * SP + xc8) = xrb;                 // ds_store_b128
    } else {
      __bf16* pa = xs + xrowA * SP + xc4a;
      pa[0] = f2bf(xrfA.x); pa[1] = f2bf(xrfA.y);
      pa[2] = f2bf(xrfA.z); pa[3] = f2bf(xrfA.w);
      __bf16* pb = xs + xrowB * SP + xc4b;
      pb[0] = f2bf(xrfB.x); pb[1] = f2bf(xrfB.y);
      pb[2] = f2bf(xrfB.z); pb[3] = f2bf(xrfB.w);
    }
  };

  fetch_x(0);
  __syncthreads();

  for (int t = 0; t < SEQ; ++t) {
    store_x();                 // land prefetched x_t into LDS
    __syncthreads();
    if (t + 1 < SEQ) fetch_x(t + 1);  // retires behind two cells of compute

    mog_cell(xs, h1s, c1, l1_mogp, l1_wihp, l1_whhp, mb1, gb1,
             wv, lane, nullptr, m0);

    // layer-2 input is a copy of h1 (mogrify must not clobber carried h1)
    for (int v = tid; v < 4096; v += 1024) {
      int row = v >> 8;
      int cc  = (v & 255) << 1;
      *(unsigned*)(xs + row * SP + cc) = *(const unsigned*)(h1s + row * SP + cc);
    }
    __syncthreads();

    mog_cell(xs, h2s, c2, l2_mogp, l2_wihp, l2_whhp, mb2, gb2,
             wv, lane, out + (size_t)t * BATCH * HD, m0);
  }

  // ---- epilogue: last_hidden [2,B,H] then last_cell [2,B,H] ----
  const size_t lh = (size_t)SEQ * BATCH * HD;
  for (int v = tid; v < 8192; v += 1024) {
    int row = v >> 9;
    int c   = v & 511;
    out[lh + (size_t)(m0 + row) * HD + c] = bf2f(h1s[row * SP + c]);
    out[lh + (size_t)BATCH * HD + (size_t)(m0 + row) * HD + c] =
        bf2f(h2s[row * SP + c]);
  }
  const size_t lc = lh + 2ul * BATCH * HD;
  const int rbase = (lane >> 4) << 3;
#pragma unroll
  for (int r = 0; r < 8; ++r) {
    int row = rbase + r;
    out[lc + (size_t)(m0 + row) * HD + col] = c1[r];
    out[lc + (size_t)BATCH * HD + (size_t)(m0 + row) * HD + col] = c2[r];
  }
}

// ---------------------------------------------------------------------------

extern "C" void kernel_launch(void* const* d_in, const int* in_sizes, int n_in,
                              void* d_out, int out_size, void* d_ws,
                              size_t ws_size, hipStream_t stream) {
  (void)in_sizes; (void)n_in; (void)out_size;

  const float* input_seq = (const float*)d_in[0];
  const float* l1_mogW   = (const float*)d_in[1];
  const float* l1_mogb   = (const float*)d_in[2];
  const float* l1_Wih    = (const float*)d_in[3];
  const float* l1_Whh    = (const float*)d_in[4];
  const float* l1_bih    = (const float*)d_in[5];
  const float* l1_bhh    = (const float*)d_in[6];
  const float* l2_mogW   = (const float*)d_in[7];
  const float* l2_mogb   = (const float*)d_in[8];
  const float* l2_Wih    = (const float*)d_in[9];
  const float* l2_Whh    = (const float*)d_in[10];
  const float* l2_bih    = (const float*)d_in[11];
  const float* l2_bhh    = (const float*)d_in[12];
  float* out = (float*)d_out;

  // workspace layout (bf16 elems): packed weights (13.6 MB), then optional
  // bf16 copy of input_seq (33.5 MB) if ws is large enough.
  __bf16* ws = (__bf16*)d_ws;
  size_t o = 0;
  __bf16* l1_mogp = ws + o; o += (size_t)NSTEP * HD * HD;   // 5 x 512x512
  __bf16* l1_wihp = ws + o; o += (size_t)4 * HD * HD;       // 2048x512
  __bf16* l1_whhp = ws + o; o += (size_t)4 * HD * HD;
  __bf16* l2_mogp = ws + o; o += (size_t)NSTEP * HD * HD;
  __bf16* l2_wihp = ws + o; o += (size_t)4 * HD * HD;
  __bf16* l2_whhp = ws + o; o += (size_t)4 * HD * HD;
  const size_t xelems = (size_t)SEQ * BATCH * HD;
  __bf16* xbf = ws + o;
  const bool use_xbf = (ws_size >= (o + xelems) * sizeof(__bf16));

  auto pack = [&](const float* src, __bf16* dst, int nrows, int ncols) {
    int total = nrows * ncols;
    pack_w_kernel<<<(total + 255) / 256, 256, 0, stream>>>(src, dst, nrows, ncols);
  };
  // stacked mogW packs cleanly as (5*512) x 512: tiles never cross matrices
  pack(l1_mogW, l1_mogp, NSTEP * HD, HD);
  pack(l1_Wih,  l1_wihp, 4 * HD, HD);
  pack(l1_Whh,  l1_whhp, 4 * HD, HD);
  pack(l2_mogW, l2_mogp, NSTEP * HD, HD);
  pack(l2_Wih,  l2_wihp, 4 * HD, HD);
  pack(l2_Whh,  l2_whhp, 4 * HD, HD);

  if (use_xbf) {
    int total4 = (int)(xelems / 4);
    cvt_x_kernel<<<(total4 + 255) / 256, 256, 0, stream>>>(input_seq, xbf, total4);
    moglstm_main<true><<<BATCH / 16, 1024, 0, stream>>>(
        input_seq, xbf, l1_mogb, l1_bih, l1_bhh, l2_mogb, l2_bih, l2_bhh,
        l1_mogp, l1_wihp, l1_whhp, l2_mogp, l2_wihp, l2_whhp, out);
  } else {
    moglstm_main<false><<<BATCH / 16, 1024, 0, stream>>>(
        input_seq, (const __bf16*)nullptr, l1_mogb, l1_bih, l1_bhh,
        l2_mogb, l2_bih, l2_bhh,
        l1_mogp, l1_wihp, l1_whhp, l2_mogp, l2_wihp, l2_whhp, out);
  }
}
